// DeformableConv2d_v3_23141283791479
// MI455X (gfx1250) — matrix operand order repaired and checked by reading, not compiled, not run on hardware
//
#include <hip/hip_runtime.h>
#include <hip/hip_bf16.h>

// ---------------- types for CDNA5 WMMA ----------------
typedef __attribute__((ext_vector_type(16))) __bf16 v16bf;
typedef __attribute__((ext_vector_type(8)))  __bf16 v8bf;
typedef __attribute__((ext_vector_type(8)))  float  v8f;
typedef __attribute__((__vector_size__(4 * sizeof(int)))) int vsi4;  // b128 payload

// optional gfx1250 async global->LDS path (probe via __has_builtin; safe fallback)
#if defined(__has_builtin)
#if __has_builtin(__builtin_amdgcn_global_load_async_to_lds_b128)
#define HAVE_ASYNC_LDS 1
#endif
#endif

#define ASG __attribute__((address_space(1)))
#define ASL __attribute__((address_space(3)))

__device__ __forceinline__ void wait_asynccnt0() {
#if defined(HAVE_ASYNC_LDS)
#if __has_builtin(__builtin_amdgcn_s_wait_asynccnt)
  __builtin_amdgcn_s_wait_asynccnt(0);
#else
  asm volatile("s_wait_asynccnt 0x0" ::: "memory");
#endif
#endif
}

// ---------------- problem constants ----------------
#define BATCH 4
#define HDIM 128
#define WDIM 128
#define CH   64          // internal channels
#define GRP  4
#define GC   16
#define PTS  9           // K*K sampling points
#define NPIX (BATCH * HDIM * WDIM)       // 65536
#define NPAD_HEADS 112                   // 72 offset + 36 mask, padded to 7 tiles

__device__ __forceinline__ v8f zero8() {
  v8f z;
#pragma unroll
  for (int i = 0; i < 8; ++i) z[i] = 0.0f;
  return z;
}

__device__ __forceinline__ v16bf zero16bf() {
  v16bf z;
#pragma unroll
  for (int i = 0; i < 16; ++i) z[i] = (__bf16)0.0f;
  return z;
}

// Build A-operand: lane holds 16 contiguous k values as two 8-element groups
// (lanes 0-15: k 0..15; lanes 16-31: k 16..31), matching packed B.
__device__ __forceinline__ v16bf make_a(const __bf16* g0, const __bf16* g1) {
  v8bf lo = *reinterpret_cast<const v8bf*>(g0);
  v8bf hi = *reinterpret_cast<const v8bf*>(g1);
  v16bf a;
#pragma unroll
  for (int i = 0; i < 8; ++i) { a[i] = lo[i]; a[8 + i] = hi[i]; }
  return a;
}

__device__ __forceinline__ v8f wmma_bf16(v16bf a, v16bf b, v8f c) {
  return __builtin_amdgcn_wmma_f32_16x16x32_bf16(
      /*neg_a=*/false, a, /*neg_b=*/false, b,
      /*c_mod=*/(short)0, c, /*reuse_a=*/false, /*reuse_b=*/false);
}

__device__ __forceinline__ float gelu_tanh(float x) {
  float x3 = x * x * x;
  return 0.5f * x * (1.0f + tanhf(0.7978845608028654f * (x + 0.044715f * x3)));
}

// cooperative stage of packed-B into LDS (16B per element per iteration)
__device__ __forceinline__ void stage_b_lds(__bf16* smem, const __bf16* Bpk, int n_v8) {
  for (int i = threadIdx.x; i < n_v8; i += blockDim.x) {
#if defined(HAVE_ASYNC_LDS)
    __builtin_amdgcn_global_load_async_to_lds_b128(
        (ASG vsi4*)(Bpk + (size_t)i * 8), (ASL vsi4*)(smem + (size_t)i * 8), 0, 0);
#else
    *reinterpret_cast<v8bf*>(smem + (size_t)i * 8) =
        *reinterpret_cast<const v8bf*>(Bpk + (size_t)i * 8);
#endif
  }
  wait_asynccnt0();
  __syncthreads();
}

// ---------------- prep kernels ----------------

// NCHW f32 -> NHWC bf16 (coalesced writes; channel = tid & 63)
__global__ void nchw_to_nhwc_bf16(const float* __restrict__ src, __bf16* __restrict__ dst) {
  size_t t = (size_t)blockIdx.x * blockDim.x + threadIdx.x;   // total NPIX*CH
  int c = (int)(t & 63);
  size_t pix = t >> 6;
  int b = (int)(pix >> 14);
  int hw = (int)(pix & 16383);
  dst[t] = (__bf16)src[(((size_t)b * 64 + c) << 14) + hw];
}

// Pack row-major [Ktot x N] f32 weights into WMMA-B tile layout, bf16,
// with column padding to Npad. dst element i: t = i&15, lane = (i>>4)&31,
// chunk index = i>>9 -> (kc, ntile); per lane: 16 contiguous k values.
__global__ void pack_b_bf16(const float* __restrict__ src, __bf16* __restrict__ dst,
                            int Ktot, int N, int Npad) {
  int i = blockIdx.x * blockDim.x + threadIdx.x;
  int total = Npad * Ktot;
  if (i >= total) return;
  int t = i & 15;
  int lane = (i >> 4) & 31;
  int idx = i >> 9;
  int nkch = Ktot >> 5;
  int kc = idx % nkch;
  int ntile = idx / nkch;
  int k = kc * 32 + ((lane >> 4) << 4) + t;
  int n = ntile * 16 + (lane & 15);
  dst[i] = (__bf16)((n < N) ? src[(size_t)k * N + n] : 0.0f);
}

// Concatenate offset (64x72) and mask (64x36) heads into packed [64 x 112] bf16.
__global__ void pack_heads_bf16(const float* __restrict__ wo, const float* __restrict__ wm,
                                __bf16* __restrict__ dst) {
  int i = blockIdx.x * blockDim.x + threadIdx.x;
  if (i >= NPAD_HEADS * 64) return;
  int t = i & 15;
  int lane = (i >> 4) & 31;
  int idx = i >> 9;
  int kc = idx & 1;           // nkch = 64/32 = 2
  int ntile = idx >> 1;
  int k = kc * 32 + ((lane >> 4) << 4) + t;
  int n = ntile * 16 + (lane & 15);
  float v = 0.0f;
  if (n < 72)       v = wo[(size_t)k * 72 + n];
  else if (n < 108) v = wm[(size_t)k * 36 + (n - 72)];
  dst[i] = (__bf16)v;
}

__global__ void build_heads_bias(const float* __restrict__ bo, const float* __restrict__ bm,
                                 float* __restrict__ dst) {
  int n = blockIdx.x * blockDim.x + threadIdx.x;
  if (n >= NPAD_HEADS) return;
  float v = 0.0f;
  if (n < 72)       v = bo[n];
  else if (n < 108) v = bm[n - 72];
  dst[n] = v;
}

// ---------------- GEMM kernels (K=64, WMMA bf16, B staged in LDS) ----------------
// One wave computes a 16 x (NT*16) tile of D = A(bf16) * Bpk(bf16) + bias.
// MODE 0: store f32 row-major [M, NT*16].
// MODE 1: store f32 scattered NCHW (out projection -> d_out).
template <int NT, int MODE>
__global__ void gemm_k64(const __bf16* __restrict__ A, const __bf16* __restrict__ Bpk,
                         const float* __restrict__ bias, float* __restrict__ out, int M) {
  extern __shared__ __bf16 smem[];                 // NT*2*32*16 bf16 = NT*2048 bytes
  stage_b_lds(smem, Bpk, NT * 128);                // NT*1024 elems / 8

  const int lane = threadIdx.x & 31;
  const int wave = threadIdx.x >> 5;
  const int tile = blockIdx.x * (blockDim.x >> 5) + wave;
  const int mbase = tile << 4;
  if (mbase >= M) return;
  const int h = lane >> 4;
  const int nl = lane & 15;
  const __bf16* arow = A + (size_t)(mbase + nl) * 64;

  v8f acc[NT];
#pragma unroll
  for (int t = 0; t < NT; ++t) acc[t] = zero8();

#pragma unroll
  for (int kc = 0; kc < 2; ++kc) {
    const int kb = kc * 32;
    v16bf a = make_a(arow + kb + 16 * h, arow + kb + 16 * h + 8);
#pragma unroll
    for (int t = 0; t < NT; ++t) {
      v16bf b = *reinterpret_cast<const v16bf*>(smem + (((size_t)t * 2 + kc) * 32 + lane) * 16);
      acc[t] = wmma_bf16(a, b, acc[t]);
    }
  }

#pragma unroll
  for (int t = 0; t < NT; ++t) {
    const int n = t * 16 + nl;
    const float bv = bias[n];
#pragma unroll
    for (int i = 0; i < 8; ++i) {
      const int m = mbase + 8 * h + i;
      const float v = acc[t][i] + bv;
      if (MODE == 0) {
        out[(size_t)m * (NT * 16) + n] = v;
      } else {
        const int b = m >> 14;
        const int hw = m & 16383;
        out[(((size_t)b * 64 + n) << 14) + hw] = v;   // NCHW
      }
    }
  }
}

// ---------------- implicit-GEMM 3x3 conv + bias + GELU -> feat bf16 ----------------
// K = 9 taps * 64 ci = 576 -> per tap, two 32-wide chunks from one 128B pixel row.
// Full packed weight matrix (72KB) staged in LDS, shared by all 8 waves.
__global__ void conv3x3_gemm(const __bf16* __restrict__ X, const __bf16* __restrict__ Bpk,
                             const float* __restrict__ bias, __bf16* __restrict__ featbf) {
  extern __shared__ __bf16 smem[];                 // 576*64 bf16 = 73728 bytes
  stage_b_lds(smem, Bpk, (576 * 64) / 8);

  const int lane = threadIdx.x & 31;
  const int wave = threadIdx.x >> 5;
  const int tile = blockIdx.x * (blockDim.x >> 5) + wave;
  const int mbase = tile << 4;
  if (mbase >= NPIX) return;
  const int h = lane >> 4;
  const int nl = lane & 15;
  const int b = mbase >> 14;
  const int rem = mbase & 16383;
  const int oy = rem >> 7;
  const int oxb = rem & 127;          // multiple of 16; tile never crosses a row
  const int ox = oxb + nl;            // this lane's pixel column

  v8f acc[4];
#pragma unroll
  for (int t = 0; t < 4; ++t) acc[t] = zero8();

#pragma unroll
  for (int tap = 0; tap < 9; ++tap) {
    const int ky = tap / 3;
    const int kx = tap - ky * 3;
    const int y = oy + ky - 1;
    const int x = ox + kx - 1;
    const bool valid = ((unsigned)y < (unsigned)HDIM) && ((unsigned)x < (unsigned)WDIM);
    v16bf a0, a1;
    if (valid) {
      const __bf16* p = X + (((size_t)b << 14) + ((size_t)y << 7) + x) * 64;
      a0 = make_a(p + 16 * h, p + 16 * h + 8);        // ci 0..31
      a1 = make_a(p + 32 + 16 * h, p + 40 + 16 * h);   // ci 32..63
    } else {
      a0 = zero16bf();
      a1 = zero16bf();
    }
#pragma unroll
    for (int t = 0; t < 4; ++t) {
      v16bf b0 = *reinterpret_cast<const v16bf*>(
          smem + (((size_t)t * 18 + tap * 2 + 0) * 32 + lane) * 16);
      acc[t] = wmma_bf16(a0, b0, acc[t]);
      v16bf b1 = *reinterpret_cast<const v16bf*>(
          smem + (((size_t)t * 18 + tap * 2 + 1) * 32 + lane) * 16);
      acc[t] = wmma_bf16(a1, b1, acc[t]);
    }
  }

#pragma unroll
  for (int t = 0; t < 4; ++t) {
    const int n = t * 16 + nl;
    const float bv = bias[n];
#pragma unroll
    for (int i = 0; i < 8; ++i) {
      const int m = mbase + 8 * h + i;
      featbf[(size_t)m * 64 + n] = (__bf16)gelu_tanh(acc[t][i] + bv);
    }
  }
}

// ---------------- deformable sampling (softmax + bilinear gather) ----------------
// thread = (pixel, group, gc); gc = lane low 4 bits -> 64B coalesced gathers.
__global__ void dcn_sample(const float* __restrict__ value, const float* __restrict__ offmask,
                           __bf16* __restrict__ dcnbf) {
  size_t t = (size_t)blockIdx.x * blockDim.x + threadIdx.x;   // total NPIX*64
  const int gc = (int)(t & 15);
  const int g = (int)((t >> 4) & 3);
  const size_t pix = t >> 6;
  const int b = (int)(pix >> 14);
  const int rem = (int)(pix & 16383);
  const int oy = rem >> 7;
  const int ox = rem & 127;

  const float* om = offmask + pix * NPAD_HEADS;

  // softmax over the 9 mask logits of this group
  float lg[PTS];
  float mx = -3.4e38f;
#pragma unroll
  for (int p = 0; p < PTS; ++p) {
    lg[p] = om[72 + g * PTS + p];
    mx = fmaxf(mx, lg[p]);
  }
  float wsum = 0.0f;
  float wgt[PTS];
#pragma unroll
  for (int p = 0; p < PTS; ++p) {
    wgt[p] = __expf(lg[p] - mx);
    wsum += wgt[p];
  }
  const float inv = 1.0f / wsum;

  const size_t bbase = ((size_t)b << 14);
  const int ch = g * 16 + gc;
  float acc = 0.0f;

#pragma unroll
  for (int p = 0; p < PTS; ++p) {
    const float offx = om[(g * PTS + p) * 2 + 0];
    const float offy = om[(g * PTS + p) * 2 + 1];
    const float px = (float)(ox + (p % 3) - 1) + offx;   // unpadded coords
    const float py = (float)(oy + (p / 3) - 1) + offy;
    const float x0f = floorf(px);
    const float y0f = floorf(py);
    const int x0 = (int)x0f;
    const int y0 = (int)y0f;
    const float fx = px - x0f;
    const float fy = py - y0f;
    float s = 0.0f;
#pragma unroll
    for (int dy = 0; dy < 2; ++dy) {
#pragma unroll
      for (int dx = 0; dx < 2; ++dx) {
        const int xi = x0 + dx;
        const int yi = y0 + dy;
        if ((unsigned)xi < (unsigned)WDIM && (unsigned)yi < (unsigned)HDIM) {
          const float wv = (dx ? fx : 1.0f - fx) * (dy ? fy : 1.0f - fy);
          s += wv * value[(bbase + ((size_t)yi << 7) + xi) * 64 + ch];
        }
      }
    }
    acc += wgt[p] * inv * s;
  }
  dcnbf[pix * 64 + ch] = (__bf16)acc;
}

// ---------------- launch ----------------
static inline size_t align256(size_t x) { return (x + 255) & ~(size_t)255; }

extern "C" void kernel_launch(void* const* d_in, const int* in_sizes, int n_in,
                              void* d_out, int out_size, void* d_ws, size_t ws_size,
                              hipStream_t stream) {
  const float* x1 = (const float*)d_in[0];
  const float* x2 = (const float*)d_in[1];
  const float* w_value = (const float*)d_in[2];
  const float* b_value = (const float*)d_in[3];
  const float* w_conv = (const float*)d_in[4];   // [3,3,64,64] == [576,64] k-major
  const float* b_conv = (const float*)d_in[5];
  const float* w_offset = (const float*)d_in[6]; // [64,72]
  const float* b_offset = (const float*)d_in[7];
  const float* w_mask = (const float*)d_in[8];   // [64,36]
  const float* b_mask = (const float*)d_in[9];
  const float* w_out = (const float*)d_in[10];
  const float* b_out = (const float*)d_in[11];
  float* out = (float*)d_out;

  // workspace carve-up
  char* w = (char*)d_ws;
  size_t off = 0;
  __bf16* x1bf = (__bf16*)(w + off); off = align256(off + (size_t)NPIX * 64 * 2);
  __bf16* x2bf = (__bf16*)(w + off); off = align256(off + (size_t)NPIX * 64 * 2);
  __bf16* featbf = (__bf16*)(w + off); off = align256(off + (size_t)NPIX * 64 * 2);
  __bf16* dcnbf = (__bf16*)(w + off); off = align256(off + (size_t)NPIX * 64 * 2);
  float* value = (float*)(w + off); off = align256(off + (size_t)NPIX * 64 * 4);
  float* offmask = (float*)(w + off); off = align256(off + (size_t)NPIX * NPAD_HEADS * 4);
  __bf16* wv_pk = (__bf16*)(w + off); off = align256(off + (size_t)64 * 64 * 2);
  __bf16* wc_pk = (__bf16*)(w + off); off = align256(off + (size_t)576 * 64 * 2);
  __bf16* wom_pk = (__bf16*)(w + off); off = align256(off + (size_t)64 * NPAD_HEADS * 2);
  __bf16* wo_pk = (__bf16*)(w + off); off = align256(off + (size_t)64 * 64 * 2);
  float* bias_om = (float*)(w + off); off = align256(off + (size_t)NPAD_HEADS * 4);
  (void)ws_size; (void)in_sizes; (void)n_in; (void)out_size;

  const int TPB = 256;
  const int elem_blocks = (NPIX * 64) / TPB;               // 16384

  // 1) NCHW -> NHWC bf16 for both streams
  nchw_to_nhwc_bf16<<<elem_blocks, TPB, 0, stream>>>(x1, x1bf);
  nchw_to_nhwc_bf16<<<elem_blocks, TPB, 0, stream>>>(x2, x2bf);

  // 2) pack weights into WMMA B layout
  pack_b_bf16<<<(64 * 64 + TPB - 1) / TPB, TPB, 0, stream>>>(w_value, wv_pk, 64, 64, 64);
  pack_b_bf16<<<(576 * 64 + TPB - 1) / TPB, TPB, 0, stream>>>(w_conv, wc_pk, 576, 64, 64);
  pack_b_bf16<<<(64 * 64 + TPB - 1) / TPB, TPB, 0, stream>>>(w_out, wo_pk, 64, 64, 64);
  pack_heads_bf16<<<(64 * NPAD_HEADS + TPB - 1) / TPB, TPB, 0, stream>>>(w_offset, w_mask, wom_pk);
  build_heads_bias<<<1, 128, 0, stream>>>(b_offset, b_mask, bias_om);

  const int gemm_blocks = (NPIX / 16) / (TPB / 32);        // 512

  // 3) value projection (f32 out for sampling gathers); B staged in LDS (8KB)
  gemm_k64<4, 0><<<gemm_blocks, TPB, 4 * 2048, stream>>>(x1bf, wv_pk, b_value, value, NPIX);

  // 4) 3x3 conv + GELU -> feat bf16; full 72KB weight tile set in LDS
  conv3x3_gemm<<<gemm_blocks, TPB, 576 * 64 * 2, stream>>>(x2bf, wc_pk, b_conv, featbf);

  // 5) offset + mask heads (raw logits; softmax fused into sampler); 14KB LDS
  gemm_k64<7, 0><<<gemm_blocks, TPB, 7 * 2048, stream>>>(featbf, wom_pk, bias_om, offmask, NPIX);

  // 6) deformable bilinear sampling -> dcn bf16
  dcn_sample<<<elem_blocks, TPB, 0, stream>>>(value, offmask, dcnbf);

  // 7) output projection, scattered straight into NCHW d_out; 8KB LDS
  gemm_k64<4, 1><<<gemm_blocks, TPB, 4 * 2048, stream>>>(dcnbf, wo_pk, b_out, out, NPIX);
}
